// PRISM_25838523253352
// MI455X (gfx1250) — matrix operand order repaired and checked by reading, not compiled, not run on hardware
//
#include <hip/hip_runtime.h>

// ---------------------------------------------------------------------------
// Problem constants (from the reference)
// ---------------------------------------------------------------------------
#define NN      16384
#define EE      262144
#define KTOP    10
#define SRCIN   2048
#define TGTIN   1024
#define HID     512
#define OUTD    64
#define FUSE    128          // 2*OUT
#define NHEAD   8
#define DH      16           // FUSE / NHEAD
#define MLPD    64
#define CHUNK   64
#define PCAK    8
#define NB      (NN * (KTOP + 1))   // 180224 transformer rows
#define DINT    (OUTD * OUTD)       // 4096 interaction features

typedef unsigned short ushort_t;
typedef __attribute__((ext_vector_type(16))) __bf16        v16bf;
typedef __attribute__((ext_vector_type(8)))  float         v8f;
typedef __attribute__((ext_vector_type(4)))  unsigned int  v4u;
typedef __attribute__((ext_vector_type(4)))  int           v4i;

union Frag { v4u u[2]; v16bf v; };

#define AS1 __attribute__((address_space(1)))
#define AS3 __attribute__((address_space(3)))

// gfx1250 async global->LDS path (ASYNCcnt), guarded so the file always builds
#if defined(__has_builtin)
#  if __has_builtin(__builtin_amdgcn_global_load_async_to_lds_b128)
#    define USE_ASYNC_LDS 1
#  endif
#endif
#ifndef USE_ASYNC_LDS
#  define USE_ASYNC_LDS 0
#endif

__device__ __forceinline__ void wait_async0() {
#if USE_ASYNC_LDS
#  if __has_builtin(__builtin_amdgcn_s_wait_asynccnt)
  __builtin_amdgcn_s_wait_asynccnt(0);
#  else
  asm volatile("s_wait_asynccnt 0x0" ::: "memory");
#  endif
#endif
}

// ---------------------------------------------------------------------------
// small device helpers
// ---------------------------------------------------------------------------
__device__ __forceinline__ ushort_t f2bf(float f) {
  unsigned u = __float_as_uint(f);
  unsigned r = u + 0x7FFFu + ((u >> 16) & 1u);       // round-to-nearest-even
  if ((u & 0x7F800000u) == 0x7F800000u) r = u;        // inf/nan passthrough
  return (ushort_t)(r >> 16);
}
__device__ __forceinline__ float bf2f(ushort_t h) {
  return __uint_as_float(((unsigned)h) << 16);
}
__device__ __forceinline__ float sigm(float x) { return 1.f / (1.f + __expf(-x)); }

// ---------------------------------------------------------------------------
// bf16 WMMA GEMM:  C(M,N) = A(M,K) @ B(K,N)  [+bias[n]] [relu]
// A,B row-major bf16 (ushort bits), C row-major f32.
// M%64==0, N%64==0, K%32==0, buffers 16B aligned.
// Block: 128 threads = 4 waves; block tile 64x64; each wave 32x32 (2x2 WMMA).
// Async path: double-buffered LDS tiles, loads for tile t+1 issued before the
// WMMA chain of tile t; s_wait_asynccnt 0 lands after compute (latency hidden).
// ---------------------------------------------------------------------------
__global__ __launch_bounds__(128)
void k_gemm(const ushort_t* __restrict__ A, const ushort_t* __restrict__ B,
            float* __restrict__ C, const float* __restrict__ bias,
            int M, int N, int K, int act)
{
#if USE_ASYNC_LDS
  __shared__ ushort_t As[2][64][40];   // row stride 80B (16B aligned)
  __shared__ ushort_t Bs[2][32][72];   // row stride 144B (16B aligned)
#else
  __shared__ ushort_t As[1][64][40];
  __shared__ ushort_t Bs[1][32][72];
#endif

  const int tid  = threadIdx.x;
  const int lane = tid & 31;
  const int wave = tid >> 5;
  const int wrow = (wave >> 1) * 32;   // wave row offset in block tile
  const int wcol = (wave & 1) * 32;    // wave col offset
  const int bm = blockIdx.x * 64;
  const int bn = blockIdx.y * 64;

  v8f acc[2][2];
#pragma unroll
  for (int i = 0; i < 2; ++i)
#pragma unroll
    for (int j = 0; j < 2; ++j)
#pragma unroll
      for (int e = 0; e < 8; ++e) acc[i][j][e] = 0.f;

  const int arow = wrow + (lane & 15);
  const int ah   = (lane >> 4) * 8;      // K half select

#if USE_ASYNC_LDS
  // ---- async double-buffered pipeline ----
  auto stage = [&](int k0, int buf) {
#pragma unroll
    for (int i = 0; i < 2; ++i) {
      int ch = tid + i * 128;                 // 0..255
      int ar = ch >> 2, ac = (ch & 3) * 8;    // A: 64 rows x 4 chunks of 8 bf16
      __builtin_amdgcn_global_load_async_to_lds_b128(
          (AS1 v4i*)(A + (size_t)(bm + ar) * K + k0 + ac),
          (AS3 v4i*)&As[buf][ar][ac], 0, 0);
      int br = ch >> 3, bc = (ch & 7) * 8;    // B: 32 rows x 8 chunks
      __builtin_amdgcn_global_load_async_to_lds_b128(
          (AS1 v4i*)(B + (size_t)(k0 + br) * N + bn + bc),
          (AS3 v4i*)&Bs[buf][br][bc], 0, 0);
    }
  };

  stage(0, 0);
  wait_async0();
  __syncthreads();

  const int nk = K / 32;
  for (int t = 0; t < nk; ++t) {
    const int cur = t & 1, nxt = cur ^ 1;
    if (t + 1 < nk) stage((t + 1) * 32, nxt);        // overlap with compute
    if (t + 2 < nk) {                                // keep L2 warm 2 tiles out
      int k2 = (t + 2) * 32;
      __builtin_prefetch(A + (size_t)(bm + (tid >> 1)) * K + k2 + (tid & 1) * 16, 0, 3);
      __builtin_prefetch(B + (size_t)(k2 + (tid >> 2)) * N + bn + (tid & 3) * 16, 0, 3);
    }

    Frag af[2], bf[2];
#pragma unroll
    for (int i = 0; i < 2; ++i) {
      int r = arow + i * 16;
      af[i].u[0] = *(const v4u*)&As[cur][r][ah];        // K = half*8 + 0..7
      af[i].u[1] = *(const v4u*)&As[cur][r][16 + ah];   // K = 16 + half*8 + 0..7
    }
#pragma unroll
    for (int j = 0; j < 2; ++j) {
      int c0 = wcol + j * 16;                           // lane = N col, elems = K (table order)
      int n = c0 + (lane & 15);
#pragma unroll
      for (int w = 0; w < 4; ++w) {
        int kk = ah + 2 * w;
        bf[j].u[0][w] = (unsigned)Bs[cur][kk][n] | ((unsigned)Bs[cur][kk + 1][n] << 16);
        bf[j].u[1][w] = (unsigned)Bs[cur][16 + kk][n] | ((unsigned)Bs[cur][17 + kk][n] << 16);
      }
    }
#pragma unroll
    for (int i = 0; i < 2; ++i)
#pragma unroll
      for (int j = 0; j < 2; ++j)
        acc[i][j] = __builtin_amdgcn_wmma_f32_16x16x32_bf16(
            false, af[i].v, false, bf[j].v, (short)0, acc[i][j], false, false);

    wait_async0();      // next tile fully in LDS (issued before compute)
    __syncthreads();
  }
#else
  // ---- fallback: register-staged single buffer ----
  for (int k0 = 0; k0 < K; k0 += 32) {
    v4u ra[2], rb[2];
#pragma unroll
    for (int i = 0; i < 2; ++i) {
      int ch = tid + i * 128;
      ra[i] = *(const v4u*)(A + (size_t)(bm + (ch >> 2)) * K + k0 + (ch & 3) * 8);
      rb[i] = *(const v4u*)(B + (size_t)(k0 + (ch >> 3)) * N + bn + (ch & 7) * 8);
    }
#pragma unroll
    for (int i = 0; i < 2; ++i) {
      int ch = tid + i * 128;
      *(v4u*)&As[0][ch >> 2][(ch & 3) * 8] = ra[i];
      *(v4u*)&Bs[0][ch >> 3][(ch & 7) * 8] = rb[i];
    }
    __syncthreads();
    if (k0 + 32 < K) {
#pragma unroll
      for (int i = 0; i < 2; ++i) {
        int ch = tid + i * 128;
        __builtin_prefetch(A + (size_t)(bm + (ch >> 2)) * K + (k0 + 32) + (ch & 3) * 8, 0, 3);
        __builtin_prefetch(B + (size_t)(k0 + 32 + (ch >> 3)) * N + bn + (ch & 7) * 8, 0, 3);
      }
    }
    Frag af[2], bf[2];
#pragma unroll
    for (int i = 0; i < 2; ++i) {
      int r = arow + i * 16;
      af[i].u[0] = *(const v4u*)&As[0][r][ah];
      af[i].u[1] = *(const v4u*)&As[0][r][16 + ah];
    }
#pragma unroll
    for (int j = 0; j < 2; ++j) {
      int c0 = wcol + j * 16;
      int n = c0 + (lane & 15);
#pragma unroll
      for (int w = 0; w < 4; ++w) {
        int kk = ah + 2 * w;
        bf[j].u[0][w] = (unsigned)Bs[0][kk][n] | ((unsigned)Bs[0][kk + 1][n] << 16);
        bf[j].u[1][w] = (unsigned)Bs[0][16 + kk][n] | ((unsigned)Bs[0][17 + kk][n] << 16);
      }
    }
#pragma unroll
    for (int i = 0; i < 2; ++i)
#pragma unroll
      for (int j = 0; j < 2; ++j)
        acc[i][j] = __builtin_amdgcn_wmma_f32_16x16x32_bf16(
            false, af[i].v, false, bf[j].v, (short)0, acc[i][j], false, false);
    __syncthreads();
  }
#endif

  // --- store C (VGPR v, lane l: M = v + (l>>4)*8, N = l&15) ---
  const int crow = (lane >> 4) * 8;
  const int ccol = lane & 15;
#pragma unroll
  for (int i = 0; i < 2; ++i)
#pragma unroll
    for (int j = 0; j < 2; ++j) {
      int n = bn + wcol + j * 16 + ccol;
      float bv = bias ? bias[n] : 0.f;
#pragma unroll
      for (int v = 0; v < 8; ++v) {
        int m = bm + wrow + i * 16 + crow + v;
        float val = acc[i][j][v] + bv;
        if (act == 1) val = fmaxf(val, 0.f);
        C[(size_t)m * N + n] = val;
      }
    }
}

// ---------------------------------------------------------------------------
// elementwise / conversion kernels
// ---------------------------------------------------------------------------
__global__ void k_f2bf(const float* __restrict__ in, ushort_t* __restrict__ out, size_t n) {
  size_t i = (size_t)blockIdx.x * blockDim.x + threadIdx.x;
  if (i < n) out[i] = f2bf(in[i]);
}
__global__ void k_trans(const float* __restrict__ in, ushort_t* __restrict__ out, int R, int C) {
  size_t i = (size_t)blockIdx.x * blockDim.x + threadIdx.x;
  if (i < (size_t)R * C) {
    int r = (int)(i / C), c = (int)(i % C);
    out[(size_t)c * R + r] = f2bf(in[i]);
  }
}
__global__ void k_elu(float* __restrict__ x, size_t n) {
  size_t i = (size_t)blockIdx.x * blockDim.x + threadIdx.x;
  if (i < n) { float v = x[i]; x[i] = v > 0.f ? v : (__expf(v) - 1.f); }
}

// ---------------------------------------------------------------------------
// node-level attention logits: out[r] = dot(X[r,:], a)  (D = 512)
// ---------------------------------------------------------------------------
__global__ __launch_bounds__(128)
void k_rowdot(const float* __restrict__ X, const float* __restrict__ a,
              float* __restrict__ out, int D) {
  int row = blockIdx.x, t = threadIdx.x;
  __shared__ float red[128];
  float p = 0.f;
  for (int d = t; d < D; d += 128) p += X[(size_t)row * D + d] * a[d];
  red[t] = p; __syncthreads();
  for (int st = 64; st > 0; st >>= 1) { if (t < st) red[t] += red[t + st]; __syncthreads(); }
  if (t == 0) out[row] = red[0];
}

// ---------------------------------------------------------------------------
// edge softmax (segment over dst) + scatter aggregation
// ---------------------------------------------------------------------------
__global__ void k_edge_max(const int* __restrict__ se, const float* __restrict__ als,
                           const float* __restrict__ ald, float* __restrict__ segmax) {
  int i = blockIdx.x * blockDim.x + threadIdx.x;
  if (i >= EE) return;
  int s = se[i], d = se[EE + i];
  float e = sigm(als[s] + ald[d]);                       // sigmoid > 0
  atomicMax((unsigned int*)&segmax[d], __float_as_uint(e));
}
__global__ void k_edge_expsum(const int* __restrict__ se, const float* __restrict__ als,
                              const float* __restrict__ ald, const float* __restrict__ segmax,
                              float* __restrict__ ex, float* __restrict__ segsum) {
  int i = blockIdx.x * blockDim.x + threadIdx.x;
  if (i >= EE) return;
  int s = se[i], d = se[EE + i];
  float e = sigm(als[s] + ald[d]);
  float xx = __expf(e - segmax[d]);
  ex[i] = xx;
  atomicAdd(&segsum[d], xx);
}
__global__ __launch_bounds__(128)
void k_edge_agg(const float* __restrict__ xs, const int* __restrict__ se,
                const float* __restrict__ ex, const float* __restrict__ segsum,
                float* __restrict__ out) {
  int e = blockIdx.x;
  int s = se[e], d = se[EE + e];
  float alpha = ex[e] / segsum[d];
  for (int c = threadIdx.x; c < HID; c += 128)
    atomicAdd(&out[(size_t)d * HID + c], xs[(size_t)s * HID + c] * alpha);
}

// ---------------------------------------------------------------------------
// fusion tensor: fusion[n, j, :] = [ j==0 ? s2[n] : s2[nb] | j==0 ? token : t2[nb] ]
// ---------------------------------------------------------------------------
__global__ __launch_bounds__(FUSE)
void k_build_fusion(const float* __restrict__ s2, const float* __restrict__ t2,
                    const int* __restrict__ topk, const int* __restrict__ nonmiss,
                    const float* __restrict__ token, float* __restrict__ fusion) {
  int n = blockIdx.x, f = threadIdx.x;
  for (int j = 0; j <= KTOP; ++j) {
    float v;
    if (j == 0) {
      v = (f < OUTD) ? s2[(size_t)n * OUTD + f] : token[f - OUTD];
    } else {
      int nb = nonmiss[(size_t)n * KTOP + (j - 1)];
      v = (f < OUTD) ? s2[(size_t)nb * OUTD + f] : t2[(size_t)nb * OUTD + (f - OUTD)];
    }
    fusion[((size_t)n * (KTOP + 1) + j) * FUSE + f] = v;
  }
}

// ---------------------------------------------------------------------------
// multi-head attention over the 64-node chunk axis, online softmax
// grid (C=256, B=11, H=8), block 64 (one thread per query position s)
// ---------------------------------------------------------------------------
__global__ __launch_bounds__(64)
void k_attn(const float* __restrict__ qkv, float* __restrict__ o) {
  int c = blockIdx.x, b = blockIdx.y, h = blockIdx.z, s = threadIdx.x;
  __shared__ float Ks[CHUNK][DH];
  __shared__ float Vs[CHUNK][DH];
  size_t row = ((size_t)(c * CHUNK + s) * (KTOP + 1) + b) * (3 * FUSE);
  float q[DH];
#pragma unroll
  for (int d = 0; d < DH; ++d) {
    q[d]     = qkv[row + h * DH + d];
    Ks[s][d] = qkv[row + FUSE + h * DH + d];
    Vs[s][d] = qkv[row + 2 * FUSE + h * DH + d];
  }
  __syncthreads();
  float m = -1e30f, l = 0.f, acc[DH];
#pragma unroll
  for (int d = 0; d < DH; ++d) acc[d] = 0.f;
  for (int t = 0; t < CHUNK; ++t) {
    float sc = 0.f;
#pragma unroll
    for (int d = 0; d < DH; ++d) sc += q[d] * Ks[t][d];
    sc *= 0.25f;                                   // 1/sqrt(DH)
    float mn = fmaxf(m, sc);
    float corr = __expf(m - mn);
    float p = __expf(sc - mn);
    l = l * corr + p;
#pragma unroll
    for (int d = 0; d < DH; ++d) acc[d] = acc[d] * corr + p * Vs[t][d];
    m = mn;
  }
  float inv = 1.f / l;
  size_t orow = ((size_t)(c * CHUNK + s) * (KTOP + 1) + b) * FUSE;
#pragma unroll
  for (int d = 0; d < DH; ++d) o[orow + h * DH + d] = acc[d] * inv;
}

// residual add + LayerNorm over FUSE=128 features
__global__ __launch_bounds__(FUSE)
void k_add_ln(const float* __restrict__ a, const float* __restrict__ d,
              const float* __restrict__ g, const float* __restrict__ bb,
              float* __restrict__ out) {
  int row = blockIdx.x, f = threadIdx.x;
  __shared__ float red[FUSE];
  size_t idx = (size_t)row * FUSE + f;
  float v = a[idx] + d[idx];
  red[f] = v; __syncthreads();
  for (int st = 64; st > 0; st >>= 1) { if (f < st) red[f] += red[f + st]; __syncthreads(); }
  float mu = red[0] * (1.f / FUSE); __syncthreads();
  float c = v - mu;
  red[f] = c * c; __syncthreads();
  for (int st = 64; st > 0; st >>= 1) { if (f < st) red[f] += red[f + st]; __syncthreads(); }
  float var = red[0] * (1.f / FUSE);
  out[idx] = c * rsqrtf(var + 1e-5f) * g[f] + bb[f];
}

__global__ __launch_bounds__(FUSE)
void k_mean_b(const float* __restrict__ x, float* __restrict__ out) {
  int n = blockIdx.x, f = threadIdx.x;
  float s = 0.f;
  for (int b = 0; b <= KTOP; ++b) s += x[((size_t)n * (KTOP + 1) + b) * FUSE + f];
  out[(size_t)n * FUSE + f] = s * (1.f / (KTOP + 1));
}

__global__ __launch_bounds__(FUSE)
void k_write_emb(const float* __restrict__ ts, const float* __restrict__ tt,
                 float* __restrict__ emb, float* __restrict__ prism) {
  int n = blockIdx.x, f = threadIdx.x;
  float v = (f < OUTD) ? ts[(size_t)n * OUTD + f] : tt[(size_t)n * OUTD + (f - OUTD)];
  emb[(size_t)n * FUSE + f] = v;
  prism[(size_t)n * (FUSE + PCAK) + f] = v;
}

// outer-product interaction features, stored bf16 both row- and col-major
__global__ __launch_bounds__(256)
void k_outer(const float* __restrict__ ts, const float* __restrict__ tt,
             ushort_t* __restrict__ inter, ushort_t* __restrict__ interT) {
  int n = blockIdx.x, t = threadIdx.x;
  __shared__ float sA[OUTD], sB[OUTD];
  if (t < OUTD) sA[t] = ts[(size_t)n * OUTD + t];
  else if (t < 2 * OUTD) sB[t - OUTD] = tt[(size_t)n * OUTD + (t - OUTD)];
  __syncthreads();
  for (int idx = t; idx < DINT; idx += 256) {
    ushort_t h = f2bf(sA[idx >> 6] * sB[idx & 63]);
    inter[(size_t)n * DINT + idx] = h;
    interT[(size_t)idx * NN + n] = h;
  }
}

// cov -= S_p * S_q / N   (S = column sums of inter, from the mu GEMM)
__global__ void k_cov_adjust(float* __restrict__ cov, const float* __restrict__ S, float invN) {
  size_t i = (size_t)blockIdx.x * blockDim.x + threadIdx.x;
  if (i < (size_t)DINT * DINT) {
    int p = (int)(i / DINT), q = (int)(i % DINT);
    cov[i] -= S[p] * S[q] * invN;
  }
}

// deterministic pseudo-random init for subspace iteration
__global__ void k_pi_init(float* __restrict__ V, int n) {
  int i = blockIdx.x * blockDim.x + threadIdx.x;
  if (i < n) {
    unsigned h = (unsigned)i * 2654435761u;
    h ^= h >> 16; h *= 2246822519u; h ^= h >> 13;
    V[i] = ((float)(h & 0xFFFFu) / 32768.f) - 1.f;
  }
}

// Y(D,8) = cov(D,D) @ V(D,8)
__global__ __launch_bounds__(256)
void k_matvec8(const float* __restrict__ A, const float* __restrict__ V,
               float* __restrict__ Y, int D) {
  int r = blockIdx.x, t = threadIdx.x;
  float acc[PCAK];
#pragma unroll
  for (int c = 0; c < PCAK; ++c) acc[c] = 0.f;
  for (int q = t; q < D; q += 256) {
    float a = A[(size_t)r * D + q];
#pragma unroll
    for (int c = 0; c < PCAK; ++c) acc[c] += a * V[(size_t)q * PCAK + c];
  }
  __shared__ float red[256];
  for (int c = 0; c < PCAK; ++c) {
    red[t] = acc[c]; __syncthreads();
    for (int st = 128; st > 0; st >>= 1) { if (t < st) red[t] += red[t + st]; __syncthreads(); }
    if (t == 0) Y[(size_t)r * PCAK + c] = red[0];
    __syncthreads();
  }
}

// modified Gram-Schmidt on 8 columns of length D (single block)
__global__ __launch_bounds__(256)
void k_gs(float* __restrict__ Y, int D) {
  int t = threadIdx.x;
  __shared__ float red[256];
  for (int c = 0; c < PCAK; ++c) {
    for (int p = 0; p < c; ++p) {
      float partial = 0.f;
      for (int d = t; d < D; d += 256) partial += Y[(size_t)d * PCAK + p] * Y[(size_t)d * PCAK + c];
      red[t] = partial; __syncthreads();
      for (int st = 128; st > 0; st >>= 1) { if (t < st) red[t] += red[t + st]; __syncthreads(); }
      float dot = red[0]; __syncthreads();
      for (int d = t; d < D; d += 256) Y[(size_t)d * PCAK + c] -= dot * Y[(size_t)d * PCAK + p];
      __syncthreads();
    }
    float partial = 0.f;
    for (int d = t; d < D; d += 256) { float v = Y[(size_t)d * PCAK + c]; partial += v * v; }
    red[t] = partial; __syncthreads();
    for (int st = 128; st > 0; st >>= 1) { if (t < st) red[t] += red[t + st]; __syncthreads(); }
    float nrm = rsqrtf(red[0] + 1e-20f); __syncthreads();
    for (int d = t; d < D; d += 256) Y[(size_t)d * PCAK + c] *= nrm;
    __syncthreads();
  }
}

// sign normalization: sgn[c] = sign of the max-|.| element of column c
__global__ __launch_bounds__(256)
void k_signs(const float* __restrict__ V, float* __restrict__ sgn, int D) {
  int c = blockIdx.x, t = threadIdx.x;
  __shared__ float bv[256];
  __shared__ int bi[256];
  float best = -1.f; int bidx = 0;
  for (int d = t; d < D; d += 256) {
    float av = fabsf(V[(size_t)d * PCAK + c]);
    if (av > best) { best = av; bidx = d; }
  }
  bv[t] = best; bi[t] = bidx; __syncthreads();
  for (int st = 128; st > 0; st >>= 1) {
    if (t < st && bv[t + st] > bv[t]) { bv[t] = bv[t + st]; bi[t] = bi[t + st]; }
    __syncthreads();
  }
  if (t == 0) sgn[c] = (V[(size_t)bi[0] * PCAK + c] >= 0.f) ? 1.f : -1.f;
}

// prism pca columns: out[n, 128+c] = sgn[c] * sum_d (inter[n,d] - mu[d]) V[d,c]
__global__ __launch_bounds__(256)
void k_pca_proj(const ushort_t* __restrict__ inter, const float* __restrict__ S,
                const float* __restrict__ V, const float* __restrict__ sgn,
                float* __restrict__ prism, float invN) {
  int n = blockIdx.x, t = threadIdx.x;
  float acc[PCAK];
#pragma unroll
  for (int c = 0; c < PCAK; ++c) acc[c] = 0.f;
  for (int d = t; d < DINT; d += 256) {
    float x = bf2f(inter[(size_t)n * DINT + d]) - S[d] * invN;
#pragma unroll
    for (int c = 0; c < PCAK; ++c) acc[c] += x * V[(size_t)d * PCAK + c];
  }
  __shared__ float red[256];
  for (int c = 0; c < PCAK; ++c) {
    red[t] = acc[c]; __syncthreads();
    for (int st = 128; st > 0; st >>= 1) { if (t < st) red[t] += red[t + st]; __syncthreads(); }
    if (t == 0) prism[(size_t)n * (FUSE + PCAK) + FUSE + c] = red[0] * sgn[c];
    __syncthreads();
  }
}

// ---------------------------------------------------------------------------
// host orchestration
// ---------------------------------------------------------------------------
extern "C" void kernel_launch(void* const* d_in, const int* in_sizes, int n_in,
                              void* d_out, int out_size, void* d_ws, size_t ws_size,
                              hipStream_t stream) {
  (void)in_sizes; (void)n_in; (void)out_size; (void)ws_size;

  // ---- inputs (setup_inputs dict order) ----
  const float* src_x   = (const float*)d_in[0];
  const float* tgt_x   = (const float*)d_in[1];
  const int*   src_e   = (const int*)d_in[2];
  const int*   tgt_e   = (const int*)d_in[3];
  const int*   topk    = (const int*)d_in[4];
  const int*   nonmiss = (const int*)d_in[5];
  const float* W_s1s = (const float*)d_in[6],  * W_s1d = (const float*)d_in[7];
  const float* a_s1s = (const float*)d_in[8],  * a_s1d = (const float*)d_in[9];
  const float* W_s2s = (const float*)d_in[10];
  const float* W_t1s = (const float*)d_in[11], * W_t1d = (const float*)d_in[12];
  const float* a_t1s = (const float*)d_in[13], * a_t1d = (const float*)d_in[14];
  const float* W_t2s = (const float*)d_in[15], * W_t2d = (const float*)d_in[16];
  const float* a_dts = (const float*)d_in[17], * a_dtd = (const float*)d_in[18];
  const float* token = (const float*)d_in[19];
  const float* in_w  = (const float*)d_in[20], * in_b  = (const float*)d_in[21];
  const float* out_w = (const float*)d_in[22], * out_b = (const float*)d_in[23];
  const float* g1 = (const float*)d_in[24], * b1 = (const float*)d_in[25];
  const float* g2 = (const float*)d_in[26], * b2 = (const float*)d_in[27];
  const float* mw1 = (const float*)d_in[28], * mb1 = (const float*)d_in[29];
  const float* mw2 = (const float*)d_in[30], * mb2 = (const float*)d_in[31];
  const float* psw1 = (const float*)d_in[32], * psb1 = (const float*)d_in[33];
  const float* psw2 = (const float*)d_in[34], * psb2 = (const float*)d_in[35];
  const float* ptw1 = (const float*)d_in[36], * ptb1 = (const float*)d_in[37];
  const float* ptw2 = (const float*)d_in[38], * ptb2 = (const float*)d_in[39];

  // ---- output regions (flat concat, all f32) ----
  float* out_s_recon  = (float*)d_out;
  float* out_t_recon  = out_s_recon + (size_t)NN * SRCIN;
  float* out_emb      = out_t_recon + (size_t)NN * TGTIN;
  float* out_prism    = out_emb + (size_t)NN * FUSE;
  float* out_src_pred = out_prism + (size_t)NN * (FUSE + PCAK);
  float* out_tgt_pred = out_src_pred + (size_t)NN * SRCIN;

  // ---- workspace arena (bump allocator, 256B aligned) ----
  char* base = (char*)d_ws;
  size_t off = 0;
  auto A16 = [&](size_t elems) -> ushort_t* {           // bf16 buffers
    ushort_t* p = (ushort_t*)(base + off);
    off = (off + elems * sizeof(ushort_t) + 255) & ~(size_t)255;
    return p;
  };
  auto A32 = [&](size_t elems) -> float* {              // f32 buffers
    float* p = (float*)(base + off);
    off = (off + elems * sizeof(float) + 255) & ~(size_t)255;
    return p;
  };

  ushort_t* srcx_b  = A16((size_t)NN * SRCIN);
  ushort_t* tgtx_b  = A16((size_t)NN * TGTIN);
  ushort_t* Ws1s_b  = A16((size_t)SRCIN * HID);
  ushort_t* Ws1d_b  = A16((size_t)SRCIN * HID);
  ushort_t* Wt1s_b  = A16((size_t)TGTIN * HID);
  ushort_t* Wt1d_b  = A16((size_t)TGTIN * HID);
  ushort_t* Ws2s_b  = A16((size_t)HID * OUTD);
  ushort_t* Wt2s_b  = A16((size_t)HID * OUTD);
  ushort_t* Ws1sT_b = A16((size_t)HID * SRCIN);
  ushort_t* Wt1sT_b = A16((size_t)HID * TGTIN);
  ushort_t* Ws2sT_b = A16((size_t)OUTD * HID);
  ushort_t* Wt2sT_b = A16((size_t)OUTD * HID);
  ushort_t* Wt2dT_b = A16((size_t)OUTD * HID);
  ushort_t* inwT_b  = A16((size_t)FUSE * 3 * FUSE);
  ushort_t* outwT_b = A16((size_t)FUSE * FUSE);
  ushort_t* mw1T_b  = A16((size_t)FUSE * MLPD);
  ushort_t* mw2T_b  = A16((size_t)MLPD * FUSE);
  ushort_t* psw1T_b = A16((size_t)FUSE * HID);
  ushort_t* psw2T_b = A16((size_t)HID * OUTD);
  ushort_t* ptw1T_b = A16((size_t)FUSE * HID);
  ushort_t* ptw2T_b = A16((size_t)HID * OUTD);
  ushort_t* nbuf512_b = A16((size_t)NN * HID);          // reused sequentially
  ushort_t* s2_b   = A16((size_t)NN * OUTD);
  ushort_t* t2_b   = A16((size_t)NN * OUTD);
  ushort_t* fusion_b = A16((size_t)NB * FUSE);
  ushort_t* x1_b   = A16((size_t)NB * FUSE);
  ushort_t* o_b    = A16((size_t)NB * FUSE);
  ushort_t* mlph_b = A16((size_t)NB * MLPD);
  ushort_t* fused_b = A16((size_t)NN * FUSE);
  ushort_t* tfs_b  = A16((size_t)NN * OUTD);
  ushort_t* tft_b  = A16((size_t)NN * OUTD);
  ushort_t* tfsT_b = A16((size_t)OUTD * NN);
  ushort_t* inter_b  = A16((size_t)NN * DINT);
  ushort_t* interT_b = A16((size_t)DINT * NN);

  float* xs_s = A32((size_t)NN * HID);
  float* xd_s = A32((size_t)NN * HID);
  float* xs_t = A32((size_t)NN * HID);
  float* xd_t = A32((size_t)NN * HID);
  float* als_s = A32(NN); float* ald_s = A32(NN);
  float* als_t = A32(NN); float* ald_t = A32(NN);
  float* alsp  = A32(NN); float* aldp  = A32(NN);
  float* segmax_s = A32(NN); float* segsum_s = A32(NN);
  float* segmax_t = A32(NN); float* segsum_t = A32(NN);
  float* segmax_p = A32(NN); float* segsum_p = A32(NN);
  float* ex_s = A32(EE); float* ex_t = A32(EE); float* ex_p = A32(EE);
  float* agg_f = A32((size_t)NN * HID);                 // reused per aggregation
  float* s2_f = A32((size_t)NN * OUTD);
  float* t2_f = A32((size_t)NN * OUTD);
  float* fusion_f = A32((size_t)NB * FUSE);
  float* qkv_f    = A32((size_t)NB * 3 * FUSE);
  float* o_f      = A32((size_t)NB * FUSE);             // attn out, then LN2 out
  float* x1_f     = A32((size_t)NB * FUSE);
  float* mlph_f   = A32((size_t)NB * MLPD);
  float* mlpo_f   = A32((size_t)NB * FUSE);
  float* fused_f  = A32((size_t)NN * FUSE);
  float* ph_f     = A32((size_t)NN * HID);
  float* tfs_f = A32((size_t)NN * OUTD);
  float* tft_f = A32((size_t)NN * OUTD);
  float* tmpa_f = A32((size_t)NN * HID);
  float* tmpb_f = A32((size_t)NN * HID);
  float* muS_f = A32(DINT);
  float* cov_f = A32((size_t)DINT * DINT);
  float* V_f = A32((size_t)DINT * PCAK);
  float* Y_f = A32((size_t)DINT * PCAK);
  float* sgn_f = A32(PCAK);

  // ---- launch helpers ----
  auto conv = [&](const float* in, ushort_t* out, size_t n) {
    k_f2bf<<<dim3((unsigned)((n + 255) / 256)), dim3(256), 0, stream>>>(in, out, n);
  };
  auto trb = [&](const float* in, ushort_t* out, int R, int C) {
    size_t n = (size_t)R * C;
    k_trans<<<dim3((unsigned)((n + 255) / 256)), dim3(256), 0, stream>>>(in, out, R, C);
  };
  auto gemm = [&](const ushort_t* Am, const ushort_t* Bm, float* Cm, const float* bias,
                  int M, int N, int K, int act) {
    k_gemm<<<dim3(M / 64, N / 64), dim3(128), 0, stream>>>(Am, Bm, Cm, bias, M, N, K, act);
  };
  auto elu = [&](float* x, size_t n) {
    k_elu<<<dim3((unsigned)((n + 255) / 256)), dim3(256), 0, stream>>>(x, n);
  };
  auto zero = [&](void* p, size_t bytes) { (void)hipMemsetAsync(p, 0, bytes, stream); };
  const dim3 egrid((EE + 255) / 256), eblk(256);

  // ---- weight / input conversions ----
  conv(src_x, srcx_b, (size_t)NN * SRCIN);
  conv(tgt_x, tgtx_b, (size_t)NN * TGTIN);
  conv(W_s1s, Ws1s_b, (size_t)SRCIN * HID);
  conv(W_s1d, Ws1d_b, (size_t)SRCIN * HID);
  conv(W_t1s, Wt1s_b, (size_t)TGTIN * HID);
  conv(W_t1d, Wt1d_b, (size_t)TGTIN * HID);
  conv(W_s2s, Ws2s_b, (size_t)HID * OUTD);
  conv(W_t2s, Wt2s_b, (size_t)HID * OUTD);
  trb(W_s1s, Ws1sT_b, SRCIN, HID);
  trb(W_t1s, Wt1sT_b, TGTIN, HID);
  trb(W_s2s, Ws2sT_b, HID, OUTD);
  trb(W_t2s, Wt2sT_b, HID, OUTD);
  trb(W_t2d, Wt2dT_b, HID, OUTD);
  trb(in_w,  inwT_b, 3 * FUSE, FUSE);
  trb(out_w, outwT_b, FUSE, FUSE);
  trb(mw1, mw1T_b, MLPD, FUSE);
  trb(mw2, mw2T_b, FUSE, MLPD);
  trb(psw1, psw1T_b, HID, FUSE);
  trb(psw2, psw2T_b, OUTD, HID);
  trb(ptw1, ptw1T_b, HID, FUSE);
  trb(ptw2, ptw2T_b, OUTD, HID);

  // ==================== source encoder + tied decoder ====================
  gemm(srcx_b, Ws1s_b, xs_s, nullptr, NN, HID, SRCIN, 0);
  gemm(srcx_b, Ws1d_b, xd_s, nullptr, NN, HID, SRCIN, 0);
  k_rowdot<<<NN, 128, 0, stream>>>(xs_s, a_s1s, als_s, HID);
  k_rowdot<<<NN, 128, 0, stream>>>(xd_s, a_s1d, ald_s, HID);
  zero(segmax_s, NN * 4); zero(segsum_s, NN * 4);
  k_edge_max<<<egrid, eblk, 0, stream>>>(src_e, als_s, ald_s, segmax_s);
  k_edge_expsum<<<egrid, eblk, 0, stream>>>(src_e, als_s, ald_s, segmax_s, ex_s, segsum_s);
  zero(agg_f, (size_t)NN * HID * 4);
  k_edge_agg<<<EE, 128, 0, stream>>>(xs_s, src_e, ex_s, segsum_s, agg_f);
  elu(agg_f, (size_t)NN * HID);
  conv(agg_f, nbuf512_b, (size_t)NN * HID);
  gemm(nbuf512_b, Ws2s_b, s2_f, nullptr, NN, OUTD, HID, 0);
  conv(s2_f, s2_b, (size_t)NN * OUTD);
  // s_recon = elu(agg(s2 @ Ws2.T)) @ Ws1.T    (tied alpha reused)
  gemm(s2_b, Ws2sT_b, tmpa_f, nullptr, NN, HID, OUTD, 0);
  zero(agg_f, (size_t)NN * HID * 4);
  k_edge_agg<<<EE, 128, 0, stream>>>(tmpa_f, src_e, ex_s, segsum_s, agg_f);
  elu(agg_f, (size_t)NN * HID);
  conv(agg_f, nbuf512_b, (size_t)NN * HID);
  gemm(nbuf512_b, Ws1sT_b, out_s_recon, nullptr, NN, SRCIN, HID, 0);

  // ==================== target encoder + tied decoder ====================
  gemm(tgtx_b, Wt1s_b, xs_t, nullptr, NN, HID, TGTIN, 0);
  gemm(tgtx_b, Wt1d_b, xd_t, nullptr, NN, HID, TGTIN, 0);
  k_rowdot<<<NN, 128, 0, stream>>>(xs_t, a_t1s, als_t, HID);
  k_rowdot<<<NN, 128, 0, stream>>>(xd_t, a_t1d, ald_t, HID);
  zero(segmax_t, NN * 4); zero(segsum_t, NN * 4);
  k_edge_max<<<egrid, eblk, 0, stream>>>(tgt_e, als_t, ald_t, segmax_t);
  k_edge_expsum<<<egrid, eblk, 0, stream>>>(tgt_e, als_t, ald_t, segmax_t, ex_t, segsum_t);
  zero(agg_f, (size_t)NN * HID * 4);
  k_edge_agg<<<EE, 128, 0, stream>>>(xs_t, tgt_e, ex_t, segsum_t, agg_f);
  elu(agg_f, (size_t)NN * HID);
  conv(agg_f, nbuf512_b, (size_t)NN * HID);
  gemm(nbuf512_b, Wt2s_b, t2_f, nullptr, NN, OUTD, HID, 0);
  conv(t2_f, t2_b, (size_t)NN * OUTD);
  gemm(t2_b, Wt2sT_b, tmpa_f, nullptr, NN, HID, OUTD, 0);
  zero(agg_f, (size_t)NN * HID * 4);
  k_edge_agg<<<EE, 128, 0, stream>>>(tmpa_f, tgt_e, ex_t, segsum_t, agg_f);
  elu(agg_f, (size_t)NN * HID);
  conv(agg_f, nbuf512_b, (size_t)NN * HID);
  gemm(nbuf512_b, Wt1sT_b, out_t_recon, nullptr, NN, TGTIN, HID, 0);

  // ==================== fusion + transformer block ====================
  k_build_fusion<<<NN, FUSE, 0, stream>>>(s2_f, t2_f, topk, nonmiss, token, fusion_f);
  conv(fusion_f, fusion_b, (size_t)NB * FUSE);
  gemm(fusion_b, inwT_b, qkv_f, in_b, NB, 3 * FUSE, FUSE, 0);
  k_attn<<<dim3(NN / CHUNK, KTOP + 1, NHEAD), dim3(64), 0, stream>>>(qkv_f, o_f);
  conv(o_f, o_b, (size_t)NB * FUSE);
  gemm(o_b, outwT_b, o_f, out_b, NB, FUSE, FUSE, 0);              // out-proj
  k_add_ln<<<NB, FUSE, 0, stream>>>(fusion_f, o_f, g1, b1, x1_f); // LN1
  conv(x1_f, x1_b, (size_t)NB * FUSE);
  gemm(x1_b, mw1T_b, mlph_f, mb1, NB, MLPD, FUSE, 1);             // MLP up + relu
  conv(mlph_f, mlph_b, (size_t)NB * MLPD);
  gemm(mlph_b, mw2T_b, mlpo_f, mb2, NB, FUSE, MLPD, 0);           // MLP down
  k_add_ln<<<NB, FUSE, 0, stream>>>(x1_f, mlpo_f, g2, b2, o_f);   // LN2 -> o_f
  k_mean_b<<<NN, FUSE, 0, stream>>>(o_f, fused_f);
  conv(fused_f, fused_b, (size_t)NN * FUSE);

  // post-fusion projection heads
  gemm(fused_b, psw1T_b, ph_f, psb1, NN, HID, FUSE, 1);
  conv(ph_f, nbuf512_b, (size_t)NN * HID);
  gemm(nbuf512_b, psw2T_b, tfs_f, psb2, NN, OUTD, HID, 0);
  gemm(fused_b, ptw1T_b, ph_f, ptb1, NN, HID, FUSE, 1);
  conv(ph_f, nbuf512_b, (size_t)NN * HID);
  gemm(nbuf512_b, ptw2T_b, tft_f, ptb2, NN, OUTD, HID, 0);
  k_write_emb<<<NN, FUSE, 0, stream>>>(tfs_f, tft_f, out_emb, out_prism);

  // ==================== interaction features + PCA ====================
  k_outer<<<NN, 256, 0, stream>>>(tfs_f, tft_f, inter_b, interT_b);
  trb(tfs_f, tfsT_b, NN, OUTD);
  conv(tfs_f, tfs_b, (size_t)NN * OUTD);
  conv(tft_f, tft_b, (size_t)NN * OUTD);
  gemm(tfsT_b, tft_b, muS_f, nullptr, OUTD, OUTD, NN, 0);           // column sums of inter
  gemm(interT_b, inter_b, cov_f, nullptr, DINT, DINT, NN, 0);       // X^T X (big WMMA GEMM)
  {
    size_t n = (size_t)DINT * DINT;
    k_cov_adjust<<<dim3((unsigned)((n + 255) / 256)), dim3(256), 0, stream>>>(
        cov_f, muS_f, 1.f / NN);
  }
  // orthogonal (subspace) iteration for top-8 eigenvectors
  k_pi_init<<<dim3((DINT * PCAK + 255) / 256), dim3(256), 0, stream>>>(V_f, DINT * PCAK);
  k_gs<<<1, 256, 0, stream>>>(V_f, DINT);
  {
    float* cur = V_f; float* nxt = Y_f;
    for (int it = 0; it < 16; ++it) {
      k_matvec8<<<DINT, 256, 0, stream>>>(cov_f, cur, nxt, DINT);
      k_gs<<<1, 256, 0, stream>>>(nxt, DINT);
      float* t = cur; cur = nxt; nxt = t;
    }
    k_signs<<<PCAK, 256, 0, stream>>>(cur, sgn_f, DINT);
    k_pca_proj<<<NN, 256, 0, stream>>>(inter_b, muS_f, cur, sgn_f, out_prism, 1.f / NN);
  }

  // ==================== prediction decoders ====================
  // src: tied attention (reuse encoder's src alpha)
  gemm(tfs_b, Ws2sT_b, tmpa_f, nullptr, NN, HID, OUTD, 0);
  zero(agg_f, (size_t)NN * HID * 4);
  k_edge_agg<<<EE, 128, 0, stream>>>(tmpa_f, src_e, ex_s, segsum_s, agg_f);
  elu(agg_f, (size_t)NN * HID);
  conv(agg_f, nbuf512_b, (size_t)NN * HID);
  gemm(nbuf512_b, Ws1sT_b, out_src_pred, nullptr, NN, SRCIN, HID, 0);
  // tgt: fresh attention with a_dt1 logits on tgt graph
  gemm(tft_b, Wt2sT_b, tmpa_f, nullptr, NN, HID, OUTD, 0);
  gemm(tft_b, Wt2dT_b, tmpb_f, nullptr, NN, HID, OUTD, 0);
  k_rowdot<<<NN, 128, 0, stream>>>(tmpa_f, a_dts, alsp, HID);
  k_rowdot<<<NN, 128, 0, stream>>>(tmpb_f, a_dtd, aldp, HID);
  zero(segmax_p, NN * 4); zero(segsum_p, NN * 4);
  k_edge_max<<<egrid, eblk, 0, stream>>>(tgt_e, alsp, aldp, segmax_p);
  k_edge_expsum<<<egrid, eblk, 0, stream>>>(tgt_e, alsp, aldp, segmax_p, ex_p, segsum_p);
  zero(agg_f, (size_t)NN * HID * 4);
  k_edge_agg<<<EE, 128, 0, stream>>>(tmpa_f, tgt_e, ex_p, segsum_p, agg_f);
  elu(agg_f, (size_t)NN * HID);
  conv(agg_f, nbuf512_b, (size_t)NN * HID);
  gemm(nbuf512_b, Wt1sT_b, out_tgt_pred, nullptr, NN, TGTIN, HID, 0);
}
